// GINe_20641612825471
// MI455X (gfx1250) — matrix operand order repaired and checked by reading, not compiled, not run on hardware
//
#include <hip/hip_runtime.h>

// ---------------- problem constants ----------------
#define N_NODES 50000
#define N_EDGES 400000
#define HDIM    100
#define NFEAT   16
#define EFEAT   8
#define LLAYERS 2

#define KP_H    128   // K padded for H-dim (100 -> 128), 4 chunks of 32
#define KP_3H   320   // K padded for 3H-dim (300 -> 320), 10 chunks of 32
#define NT_H    7     // N tiles for H outputs (100 -> 112)
#define NT_M    4     // N tiles for 50 outputs (50 -> 64)
#define KT_H    4
#define KT_3H   10

#define BN_INV  0.9999950000375f   // 1/sqrt(1 + 1e-5)

typedef __attribute__((ext_vector_type(16))) _Float16 v16h;
typedef __attribute__((ext_vector_type(8)))  _Float16 v8h;
typedef __attribute__((ext_vector_type(2)))  _Float16 v2h;
typedef __attribute__((ext_vector_type(8)))  float    v8f;

// ---------------- WMMA helpers ----------------
__device__ __forceinline__ v8f wmma_f16(v16h a, v16h b, v8f c) {
  return __builtin_amdgcn_wmma_f32_16x16x32_f16(false, a, false, b, (short)0, c,
                                                false, false);
}

// A fragment from LDS tile [16][ldk] (row-major f16).
// ISA 16-bit A layout: lane L holds row M=L%16; lane group 0 -> K {0..7,16..23},
// group 1 -> K {8..15,24..31} within a 32-wide K chunk.
__device__ __forceinline__ v16h load_a_frag(const _Float16* A, int ldk, int kt, int lane) {
  int row = lane & 15;
  int kb  = kt * 32 + ((lane >> 4) << 3);
  const _Float16* p = A + row * ldk + kb;
  v8h lo = *(const v8h*)(p);
  v8h hi = *(const v8h*)(p + 16);
  return __builtin_shufflevector(lo, hi, 0,1,2,3,4,5,6,7,8,9,10,11,12,13,14,15);
}

// B fragment from pre-packed global buffer: 32 bytes contiguous per lane.
__device__ __forceinline__ v16h load_b_frag(const _Float16* P, int kt, int nt,
                                            int NT, int lane) {
  const _Float16* src = P + ((size_t)((kt * NT + nt) * 32 + lane) << 4);
  return *(const v16h*)src;
}

__device__ __forceinline__ v2h pack2(float a, float b) {
  v2h r; r[0] = (_Float16)a; r[1] = (_Float16)b; return r;
}

// ---------------- small elementwise kernels ----------------
__global__ void embed_nodes_kernel(const float* __restrict__ x,
                                   const float* __restrict__ w,
                                   const float* __restrict__ b,
                                   float* __restrict__ h) {
  int idx = blockIdx.x * blockDim.x + threadIdx.x;
  if (idx >= N_NODES * HDIM) return;
  int n = idx / HDIM, k = idx % HDIM;
  float s = b[k];
#pragma unroll
  for (int j = 0; j < NFEAT; ++j) s += x[n * NFEAT + j] * w[j * HDIM + k];
  h[idx] = s;
}

__global__ void embed_edges_kernel(const float* __restrict__ ea,
                                   const float* __restrict__ w,
                                   const float* __restrict__ b,
                                   float* __restrict__ e) {
  int idx = blockIdx.x * blockDim.x + threadIdx.x;
  if (idx >= N_EDGES * HDIM) return;
  int n = idx / HDIM, k = idx % HDIM;
  float s = b[k];
#pragma unroll
  for (int j = 0; j < EFEAT; ++j) s += ea[n * EFEAT + j] * w[j * HDIM + k];
  e[idx] = s;
}

__global__ void copy4_kernel(const float4* __restrict__ a, float4* __restrict__ b, int n4) {
  int idx = blockIdx.x * blockDim.x + threadIdx.x;
  if (idx < n4) b[idx] = a[idx];
}

// Pack a row-major f32 weight [K, Ncols] into WMMA B-fragment order (f16).
// ISA 16-bit B layout: lane = column (lane&15), lane group g holds K {8g..8g+7, 16+8g..16+8g+7}.
__global__ void pack_weight_kernel(const float* __restrict__ W, _Float16* __restrict__ P,
                                   int K, int Ncols, int KT, int NT) {
  int idx = blockIdx.x * blockDim.x + threadIdx.x;
  if (idx >= KT * NT * 32) return;
  int lane = idx & 31;
  int tile = idx >> 5;
  int nt = tile % NT, kt = tile / NT;
  int col = nt * 16 + (lane & 15);
  int kb  = kt * 32 + ((lane >> 4) << 3);
  _Float16* dst = P + ((size_t)idx << 4);
#pragma unroll
  for (int j = 0; j < 16; ++j) {
    int k = kb + (j < 8 ? j : j + 8);
    float v = (k < K && col < Ncols) ? W[k * Ncols + col] : 0.f;
    dst[j] = (_Float16)v;
  }
}

// msg = relu(h[src] + e); agg[dst] += msg  (agg pre-initialized to h -> agg == z)
// One thread per (edge, 4 features): float4 reads, 4 atomics.
__global__ void scatter_kernel(const float* __restrict__ h, const float* __restrict__ e,
                               const int* __restrict__ src, const int* __restrict__ dst,
                               float* __restrict__ agg) {
  int idx = blockIdx.x * blockDim.x + threadIdx.x;
  if (idx >= N_EDGES * (HDIM / 4)) return;
  int edge = idx / (HDIM / 4);
  int q    = (idx % (HDIM / 4)) * 4;
  int s = src[edge], d = dst[edge];
  float4 ev = *(const float4*)(e + (size_t)edge * HDIM + q);
  float4 hv = *(const float4*)(h + (size_t)s * HDIM + q);
  float* ap = agg + (size_t)d * HDIM + q;
  float m0 = hv.x + ev.x; m0 = m0 > 0.f ? m0 : 0.f;
  float m1 = hv.y + ev.y; m1 = m1 > 0.f ? m1 : 0.f;
  float m2 = hv.z + ev.z; m2 = m2 > 0.f ? m2 : 0.f;
  float m3 = hv.w + ev.w; m3 = m3 > 0.f ? m3 : 0.f;
  atomicAdd(ap + 0, m0);
  atomicAdd(ap + 1, m1);
  atomicAdd(ap + 2, m2);
  atomicAdd(ap + 3, m3);
}

// Coalesced gather of concat(h[src], h[dst], e) into a [16][KP_3H] f16 LDS tile.
// Whole wave walks one row: lane pairs of consecutive features -> fully coalesced
// float2 loads + packed 2xf16 stores.
__device__ __forceinline__ void gather_concat_tile(
    const float* __restrict__ h, const float* __restrict__ e,
    const int* __restrict__ src, const int* __restrict__ dst,
    int ebase, int lane, _Float16* A) {
#pragma unroll 1
  for (int row = 0; row < 16; ++row) {
    int edge = ebase + row;
    bool valid = edge < N_EDGES;
    int si = valid ? src[edge] : 0;
    int di = valid ? dst[edge] : 0;
    const float* hs = h + (size_t)si * HDIM;
    const float* hd = h + (size_t)di * HDIM;
    const float* ee = e + (size_t)edge * HDIM;
    _Float16* Ar = A + row * KP_3H;
    for (int k = 2 * lane; k < HDIM; k += 64) {
      float2 a = valid ? *(const float2*)(hs + k) : make_float2(0.f, 0.f);
      float2 b = valid ? *(const float2*)(hd + k) : make_float2(0.f, 0.f);
      float2 c = valid ? *(const float2*)(ee + k) : make_float2(0.f, 0.f);
      *(v2h*)(Ar + k)            = pack2(a.x, a.y);
      *(v2h*)(Ar + HDIM + k)     = pack2(b.x, b.y);
      *(v2h*)(Ar + 2 * HDIM + k) = pack2(c.x, c.y);
    }
    for (int k = 3 * HDIM + lane; k < KP_3H; k += 32) Ar[k] = (_Float16)0.f;
  }
}

// ---------------- node MLP (WMMA) ----------------
#define NODE_WAVES 4
__global__ __launch_bounds__(NODE_WAVES * 32)
void node_mlp_kernel(const float* __restrict__ zin, float* __restrict__ h,
                     const _Float16* __restrict__ w1p, const float* __restrict__ b1,
                     const _Float16* __restrict__ w2p, const float* __restrict__ b2,
                     const float* __restrict__ gamma, const float* __restrict__ beta) {
  __shared__ __align__(32) _Float16 ldsA[NODE_WAVES][16 * KP_H];
  __shared__ __align__(32) _Float16 ldsHid[NODE_WAVES][16 * KP_H];
  int wave = threadIdx.x >> 5;
  int lane = threadIdx.x & 31;
  int mbase = (blockIdx.x * NODE_WAVES + wave) * 16;
  _Float16* A  = ldsA[wave];
  _Float16* Hd = ldsHid[wave];

  // coalesced per-row fill (lane = feature pair 2*lane, 2*lane+1)
#pragma unroll 1
  for (int row = 0; row < 16; ++row) {
    int node = mbase + row;
    bool valid = node < N_NODES;
    const float* zr = zin + (size_t)node * HDIM;
    _Float16* Ar = A + row * KP_H;
    for (int k = 2 * lane; k < HDIM; k += 64) {
      float2 a = valid ? *(const float2*)(zr + k) : make_float2(0.f, 0.f);
      *(v2h*)(Ar + k) = pack2(a.x, a.y);
    }
    for (int k = HDIM + lane; k < KP_H; k += 32) Ar[k] = (_Float16)0.f;
  }
  for (int i = lane; i < 16 * KP_H; i += 32) Hd[i] = (_Float16)0.f;
  __syncthreads();

  // hidden = relu(z @ w1 + b1)
  for (int nt = 0; nt < NT_H; ++nt) {
    v8f acc = {};
#pragma unroll
    for (int kt = 0; kt < KT_H; ++kt)
      acc = wmma_f16(load_a_frag(A, KP_H, kt, lane), load_b_frag(w1p, kt, nt, NT_H, lane), acc);
    int col  = nt * 16 + (lane & 15);
    float bias = (col < HDIM) ? b1[col] : 0.f;
    int roff = (lane >> 4) * 8;
#pragma unroll
    for (int r = 0; r < 8; ++r) {
      float v = acc[r] + bias;
      Hd[(r + roff) * KP_H + col] = (_Float16)(v > 0.f ? v : 0.f);
    }
  }
  __syncthreads();

  // z2 = hidden @ w2 + b2 ; bn ; h = (h + relu(bn)) * 0.5
  for (int nt = 0; nt < NT_H; ++nt) {
    v8f acc = {};
#pragma unroll
    for (int kt = 0; kt < KT_H; ++kt)
      acc = wmma_f16(load_a_frag(Hd, KP_H, kt, lane), load_b_frag(w2p, kt, nt, NT_H, lane), acc);
    int col = nt * 16 + (lane & 15);
    if (col < HDIM) {
      float bias = b2[col];
      float g = gamma[col] * BN_INV;
      float bt = beta[col];
      int roff = (lane >> 4) * 8;
#pragma unroll
      for (int r = 0; r < 8; ++r) {
        int node = mbase + r + roff;
        if (node < N_NODES) {
          float bn = (acc[r] + bias) * g + bt;
          float rl = bn > 0.f ? bn : 0.f;
          float hv = h[node * HDIM + col];
          h[node * HDIM + col] = (hv + rl) * 0.5f;
        }
      }
    }
  }
}

// ---------------- edge MLP (WMMA) ----------------
#define EDGE_WAVES 4
__global__ __launch_bounds__(EDGE_WAVES * 32)
void edge_mlp_kernel(const float* __restrict__ h, float* __restrict__ e,
                     const int* __restrict__ src, const int* __restrict__ dst,
                     const _Float16* __restrict__ w1p, const float* __restrict__ b1,
                     const _Float16* __restrict__ w2p, const float* __restrict__ b2) {
  __shared__ __align__(32) _Float16 ldsA[EDGE_WAVES][16 * KP_3H];
  __shared__ __align__(32) _Float16 ldsHid[EDGE_WAVES][16 * KP_H];
  int wave = threadIdx.x >> 5;
  int lane = threadIdx.x & 31;
  int ebase = (blockIdx.x * EDGE_WAVES + wave) * 16;
  _Float16* A  = ldsA[wave];
  _Float16* Hd = ldsHid[wave];

  gather_concat_tile(h, e, src, dst, ebase, lane, A);
  for (int i = lane; i < 16 * KP_H; i += 32) Hd[i] = (_Float16)0.f;
  __syncthreads();

  // hidden = relu(ei @ em_w1 + b1)
  for (int nt = 0; nt < NT_H; ++nt) {
    v8f acc = {};
#pragma unroll
    for (int kt = 0; kt < KT_3H; ++kt)
      acc = wmma_f16(load_a_frag(A, KP_3H, kt, lane), load_b_frag(w1p, kt, nt, NT_H, lane), acc);
    int col = nt * 16 + (lane & 15);
    float bias = (col < HDIM) ? b1[col] : 0.f;
    int roff = (lane >> 4) * 8;
#pragma unroll
    for (int r = 0; r < 8; ++r) {
      float v = acc[r] + bias;
      Hd[(r + roff) * KP_H + col] = (_Float16)(v > 0.f ? v : 0.f);
    }
  }
  __syncthreads();

  // e += 0.5 * (hidden @ em_w2 + b2)
  for (int nt = 0; nt < NT_H; ++nt) {
    v8f acc = {};
#pragma unroll
    for (int kt = 0; kt < KT_H; ++kt)
      acc = wmma_f16(load_a_frag(Hd, KP_H, kt, lane), load_b_frag(w2p, kt, nt, NT_H, lane), acc);
    int col = nt * 16 + (lane & 15);
    if (col < HDIM) {
      float bias = b2[col];
      int roff = (lane >> 4) * 8;
#pragma unroll
      for (int r = 0; r < 8; ++r) {
        int edge = ebase + r + roff;
        if (edge < N_EDGES) {
          float ev = e[(size_t)edge * HDIM + col];
          e[(size_t)edge * HDIM + col] = ev + 0.5f * (acc[r] + bias);
        }
      }
    }
  }
}

// ---------------- final MLP ----------------
#define FIN_WAVES 4
__global__ __launch_bounds__(FIN_WAVES * 32)
void final_mlp_kernel(const float* __restrict__ h, const float* __restrict__ e,
                      const int* __restrict__ src, const int* __restrict__ dst,
                      const _Float16* __restrict__ w1p, const float* __restrict__ b1,
                      const float* __restrict__ w2, const float* __restrict__ b2,
                      const float* __restrict__ w3, const float* __restrict__ b3,
                      float* __restrict__ out) {
  __shared__ __align__(32) _Float16 ldsA[FIN_WAVES][16 * KP_3H];
  __shared__ __align__(32) float    ldsH1[FIN_WAVES][16 * 64];
  int wave = threadIdx.x >> 5;
  int lane = threadIdx.x & 31;
  int ebase = (blockIdx.x * FIN_WAVES + wave) * 16;
  _Float16* A = ldsA[wave];
  float* H1 = ldsH1[wave];

  gather_concat_tile(h, e, src, dst, ebase, lane, A);
  __syncthreads();

  // h1 = relu(xe @ mlp_w1 + b1)   (50 valid cols, padded to 64)
  for (int nt = 0; nt < NT_M; ++nt) {
    v8f acc = {};
#pragma unroll
    for (int kt = 0; kt < KT_3H; ++kt)
      acc = wmma_f16(load_a_frag(A, KP_3H, kt, lane), load_b_frag(w1p, kt, nt, NT_M, lane), acc);
    int col = nt * 16 + (lane & 15);
    float bias = (col < 50) ? b1[col] : 0.f;
    int roff = (lane >> 4) * 8;
#pragma unroll
    for (int r = 0; r < 8; ++r) {
      float v = acc[r] + bias;
      H1[(r + roff) * 64 + col] = v > 0.f ? v : 0.f;
    }
  }
  __syncthreads();

  // scalar tail: 50 -> 25 -> 1
  if (lane < 16) {
    int edge = ebase + lane;
    if (edge < N_EDGES) {
      const float* h1 = H1 + lane * 64;
      float o3 = b3[0];
      for (int j = 0; j < 25; ++j) {
        float s = b2[j];
        for (int k = 0; k < 50; ++k) s += h1[k] * w2[k * 25 + j];
        s = s > 0.f ? s : 0.f;
        o3 += s * w3[j];
      }
      out[edge] = o3;
    }
  }
}

// ---------------- host ----------------
static constexpr size_t SZ_H   = (size_t)N_NODES * HDIM * 4;   // 20 MB
static constexpr size_t SZ_E   = (size_t)N_EDGES * HDIM * 4;   // 160 MB
static constexpr size_t CONVP  = (size_t)KT_H  * NT_H * 32 * 16 * 2;  // 28672 B
static constexpr size_t EMW1P  = (size_t)KT_3H * NT_H * 32 * 16 * 2;  // 71680 B
static constexpr size_t MLP1P  = (size_t)KT_3H * NT_M * 32 * 16 * 2;  // 40960 B

static constexpr size_t OFF_H    = 0;
static constexpr size_t OFF_AGG  = OFF_H + SZ_H;
static constexpr size_t OFF_E    = OFF_AGG + SZ_H;
static constexpr size_t OFF_CW1P = OFF_E + SZ_E;
static constexpr size_t OFF_CW2P = OFF_CW1P + LLAYERS * CONVP;
static constexpr size_t OFF_EW1P = OFF_CW2P + LLAYERS * CONVP;
static constexpr size_t OFF_EW2P = OFF_EW1P + LLAYERS * EMW1P;
static constexpr size_t OFF_MW1P = OFF_EW2P + LLAYERS * CONVP;

extern "C" void kernel_launch(void* const* d_in, const int* in_sizes, int n_in,
                              void* d_out, int out_size, void* d_ws, size_t ws_size,
                              hipStream_t stream) {
  (void)in_sizes; (void)n_in; (void)out_size; (void)ws_size;
  const float* x        = (const float*)d_in[0];
  const int*   eidx     = (const int*)d_in[1];
  const float* eattr    = (const float*)d_in[2];
  const float* node_w   = (const float*)d_in[3];
  const float* node_b   = (const float*)d_in[4];
  const float* edge_w   = (const float*)d_in[5];
  const float* edge_b   = (const float*)d_in[6];
  const float* conv_w1  = (const float*)d_in[7];
  const float* conv_b1  = (const float*)d_in[8];
  const float* conv_w2  = (const float*)d_in[9];
  const float* conv_b2  = (const float*)d_in[10];
  const float* bn_gamma = (const float*)d_in[11];
  const float* bn_beta  = (const float*)d_in[12];
  const float* em_w1    = (const float*)d_in[13];
  const float* em_b1    = (const float*)d_in[14];
  const float* em_w2    = (const float*)d_in[15];
  const float* em_b2    = (const float*)d_in[16];
  const float* mlp_w1   = (const float*)d_in[17];
  const float* mlp_b1   = (const float*)d_in[18];
  const float* mlp_w2   = (const float*)d_in[19];
  const float* mlp_b2   = (const float*)d_in[20];
  const float* mlp_w3   = (const float*)d_in[21];
  const float* mlp_b3   = (const float*)d_in[22];

  const int* srcI = eidx;
  const int* dstI = eidx + N_EDGES;

  char* ws = (char*)d_ws;
  float* h   = (float*)(ws + OFF_H);
  float* agg = (float*)(ws + OFF_AGG);
  float* e   = (float*)(ws + OFF_E);
  _Float16* cw1p = (_Float16*)(ws + OFF_CW1P);
  _Float16* cw2p = (_Float16*)(ws + OFF_CW2P);
  _Float16* ew1p = (_Float16*)(ws + OFF_EW1P);
  _Float16* ew2p = (_Float16*)(ws + OFF_EW2P);
  _Float16* mw1p = (_Float16*)(ws + OFF_MW1P);

  auto cdiv = [](long a, long b) { return (int)((a + b - 1) / b); };

  // input embeddings
  embed_nodes_kernel<<<cdiv((long)N_NODES * HDIM, 256), 256, 0, stream>>>(x, node_w, node_b, h);
  embed_edges_kernel<<<cdiv((long)N_EDGES * HDIM, 256), 256, 0, stream>>>(eattr, edge_w, edge_b, e);

  // pack weights to WMMA fragment order (tiny; done every call, deterministic)
  auto pack = [&](const float* W, _Float16* P, int K, int Nc, int KT, int NT) {
    int total = KT * NT * 32;
    pack_weight_kernel<<<cdiv(total, 256), 256, 0, stream>>>(W, P, K, Nc, KT, NT);
  };
  for (int i = 0; i < LLAYERS; ++i) {
    pack(conv_w1 + (size_t)i * HDIM * HDIM, cw1p + i * (CONVP / 2), HDIM, HDIM, KT_H, NT_H);
    pack(conv_w2 + (size_t)i * HDIM * HDIM, cw2p + i * (CONVP / 2), HDIM, HDIM, KT_H, NT_H);
    pack(em_w1 + (size_t)i * 3 * HDIM * HDIM, ew1p + i * (EMW1P / 2), 3 * HDIM, HDIM, KT_3H, NT_H);
    pack(em_w2 + (size_t)i * HDIM * HDIM, ew2p + i * (CONVP / 2), HDIM, HDIM, KT_H, NT_H);
  }
  pack(mlp_w1, mw1p, 3 * HDIM, 50, KT_3H, NT_M);

  const int nodeBlocks = cdiv(N_NODES, 16 * NODE_WAVES);
  const int edgeBlocks = cdiv(N_EDGES, 16 * EDGE_WAVES);

  for (int i = 0; i < LLAYERS; ++i) {
    // agg = h ; agg += relu(h[src] + e) scattered to dst  -> agg == z
    copy4_kernel<<<cdiv((long)N_NODES * HDIM / 4, 256), 256, 0, stream>>>(
        (const float4*)h, (float4*)agg, N_NODES * HDIM / 4);
    scatter_kernel<<<cdiv((long)N_EDGES * (HDIM / 4), 256), 256, 0, stream>>>(
        h, e, srcI, dstI, agg);
    // node update (WMMA GEMMs + BN + residual)
    node_mlp_kernel<<<nodeBlocks, NODE_WAVES * 32, 0, stream>>>(
        agg, h,
        cw1p + i * (CONVP / 2), conv_b1 + i * HDIM,
        cw2p + i * (CONVP / 2), conv_b2 + i * HDIM,
        bn_gamma + i * HDIM, bn_beta + i * HDIM);
    // edge update (WMMA GEMMs over concat)
    edge_mlp_kernel<<<edgeBlocks, EDGE_WAVES * 32, 0, stream>>>(
        h, e, srcI, dstI,
        ew1p + i * (EMW1P / 2), em_b1 + i * HDIM,
        ew2p + i * (CONVP / 2), em_b2 + i * HDIM);
  }

  final_mlp_kernel<<<edgeBlocks, FIN_WAVES * 32, 0, stream>>>(
      h, e, srcI, dstI, mw1p, mlp_b1, mlp_w2, mlp_b2, mlp_w3, mlp_b3, (float*)d_out);
}
